// QEncoder_49203145343589
// MI455X (gfx1250) — matrix operand order repaired and checked by reading, not compiled, not run on hardware
//
#include <hip/hip_runtime.h>

// ---------------------------------------------------------------------------
// QEncoder for MI455X (gfx1250): bf16 WMMA GEMM pipeline + VQ argmin
//   - weights pre-transposed during f32->bf16 convert so B staging is b128
//   - 128x128 block tile, 64x32 wave tile (8 WMMA per wave per k-step)
// ---------------------------------------------------------------------------

typedef __bf16 v16bf __attribute__((ext_vector_type(16)));
typedef float  v8f   __attribute__((ext_vector_type(8)));

static constexpr int N_ROWS  = 16384;
static constexpr int D_IN    = 1024;
static constexpr int D_H     = 4096;
static constexpr int D_Z     = 256;
static constexpr int K_CODES = 4096;

#define DEV __device__ __forceinline__

// round-to-nearest-even f32 -> bf16 bits
DEV unsigned short f2bf(float f) {
    union { float f; unsigned u; } v; v.f = f;
    unsigned r = v.u + 0x7FFFu + ((v.u >> 16) & 1u);
    return (unsigned short)(r >> 16);
}

DEV unsigned long long shflxor64(unsigned long long v, int m) {
    unsigned lo = (unsigned)v, hi = (unsigned)(v >> 32);
    lo = __shfl_xor(lo, m, 32);
    hi = __shfl_xor(hi, m, 32);
    return ((unsigned long long)hi << 32) | lo;
}

// ---------------------------------------------------------------------------
// f32 -> bf16 bulk convert (4 elems/thread, all sizes divisible by 4)
// ---------------------------------------------------------------------------
__global__ void convert_f32_bf16(const float* __restrict__ in,
                                 unsigned short* __restrict__ out, int n4) {
    int i = blockIdx.x * blockDim.x + threadIdx.x;
    if (i >= n4) return;
    float4 f = reinterpret_cast<const float4*>(in)[i];
    uint2 p;
    p.x = (unsigned)f2bf(f.x) | ((unsigned)f2bf(f.y) << 16);
    p.y = (unsigned)f2bf(f.z) | ((unsigned)f2bf(f.w) << 16);
    reinterpret_cast<uint2*>(out)[i] = p;
}

// ---------------------------------------------------------------------------
// f32 [Kd,Nc] -> bf16 transposed [Nc,Kd]; 32x32 LDS tiles, 256 threads
// ---------------------------------------------------------------------------
__global__ __launch_bounds__(256)
void transpose_convert_kernel(const float* __restrict__ W,
                              unsigned short* __restrict__ WT, int Kd, int Nc) {
    __shared__ float tile[32][33];
    const int n0 = blockIdx.x * 32, k0 = blockIdx.y * 32;
    const int tx = threadIdx.x & 31, ty = threadIdx.x >> 5;   // 32 x 8
    #pragma unroll
    for (int j = 0; j < 4; ++j)
        tile[ty + 8 * j][tx] = W[(size_t)(k0 + ty + 8 * j) * Nc + n0 + tx];
    __syncthreads();
    #pragma unroll
    for (int j = 0; j < 4; ++j)
        WT[(size_t)(n0 + ty + 8 * j) * Kd + k0 + tx] = f2bf(tile[tx][ty + 8 * j]);
}

// ---------------------------------------------------------------------------
// Fragment loaders (LDS, padded row stride 40 bf16 = 80B)
//   A frag (16x32): lane L -> row (L&15); kb=(L>=16)?8:0 ; b128 at +0,+16 elems
//   B frag (32x16): lane L -> col (L&15); kb=(L>=16)?8:0 ; b128 at +0,+16 elems
// ---------------------------------------------------------------------------
DEV v16bf load_fragA(const unsigned short* p) {
    union { int4 i[2]; v16bf v; } u;
    u.i[0] = *reinterpret_cast<const int4*>(p);
    u.i[1] = *reinterpret_cast<const int4*>(p + 16);
    return u.v;
}
DEV v16bf load_fragB(const unsigned short* p) {
    union { int4 i[2]; v16bf v; } u;
    u.i[0] = *reinterpret_cast<const int4*>(p);
    u.i[1] = *reinterpret_cast<const int4*>(p + 16);
    return u.v;
}

// ---------------------------------------------------------------------------
// GEMM: C[M,Nc] = act(A[M,Kd] @ BT[Nc,Kd]^T + bias), bf16 in, f32 accum.
// Block tile 128x128, BK=32; 8 waves in 2(M) x 4(N); wave tile 64x32.
// ---------------------------------------------------------------------------
template<bool RELU, bool STORE_F32>
__global__ __launch_bounds__(256)
void gemm_bf16_kernel(const unsigned short* __restrict__ A,
                      const unsigned short* __restrict__ BT,
                      const float* __restrict__ bias,
                      unsigned short* __restrict__ Cbf,
                      float* __restrict__ Cf,
                      int M, int Kd, int Nc) {
    __shared__ __align__(16) unsigned short sA[128 * 40];
    __shared__ __align__(16) unsigned short sB[128 * 40];

    const int tid   = threadIdx.x;
    const int lane  = tid & 31;
    const int wave  = tid >> 5;
    const int waveM = wave >> 2;   // 0..1 -> 64 rows
    const int waveN = wave & 3;    // 0..3 -> 32 cols
    const int m0 = blockIdx.y * 128;
    const int n0 = blockIdx.x * 128;

    v8f acc[4][2] = {};

    const int arow = lane & 15;
    const int kbA  = (lane >= 16) ? 8 : 0;
    const int bcol = lane & 15;
    const int kbB  = (lane >= 16) ? 8 : 0;

    // staging indices (constant across k-steps)
    const int be = tid >> 1;            // 0..127 : B entry (output col)
    const int bk = (tid & 1) * 16;      // 0 / 16 : k offset

    for (int k0 = 0; k0 < Kd; k0 += 32) {
        // stage A tile: 128 rows x 32 k, 2 x b128 per thread
        int idx = tid * 8;
        #pragma unroll
        for (int it = 0; it < 2; ++it) {
            int row = idx >> 5, kc = idx & 31;
            int4 v = *reinterpret_cast<const int4*>(
                &A[(size_t)(m0 + row) * Kd + k0 + kc]);
            *reinterpret_cast<int4*>(&sA[row * 40 + kc]) = v;
            idx += 2048;
        }
        // stage B tile: 128 entries x 32 k from transposed weights, 2 x b128
        {
            const unsigned short* g = &BT[(size_t)(n0 + be) * Kd + k0 + bk];
            int4 v0 = *reinterpret_cast<const int4*>(g);
            int4 v1 = *reinterpret_cast<const int4*>(g + 8);
            *reinterpret_cast<int4*>(&sB[be * 40 + bk])     = v0;
            *reinterpret_cast<int4*>(&sB[be * 40 + bk + 8]) = v1;
        }
        __syncthreads();

        v16bf af[4], bfr[2];
        #pragma unroll
        for (int mi = 0; mi < 4; ++mi)
            af[mi] = load_fragA(&sA[(waveM * 64 + mi * 16 + arow) * 40 + kbA]);
        #pragma unroll
        for (int ni = 0; ni < 2; ++ni)
            bfr[ni] = load_fragB(&sB[(waveN * 32 + ni * 16 + bcol) * 40 + kbB]);

        #pragma unroll
        for (int mi = 0; mi < 4; ++mi)
            #pragma unroll
            for (int ni = 0; ni < 2; ++ni)
                acc[mi][ni] = __builtin_amdgcn_wmma_f32_16x16x32_bf16(
                    false, af[mi], false, bfr[ni], (short)0, acc[mi][ni],
                    false, false);
        __syncthreads();
    }

    // epilogue: bias (+ReLU), store bf16 (and f32 for the z layer)
    const int rofs = (lane >= 16) ? 8 : 0;
    #pragma unroll
    for (int ni = 0; ni < 2; ++ni) {
        int col = n0 + waveN * 32 + ni * 16 + (lane & 15);
        float bv = bias[col];
        #pragma unroll
        for (int mi = 0; mi < 4; ++mi) {
            #pragma unroll
            for (int r = 0; r < 8; ++r) {
                int row = m0 + waveM * 64 + mi * 16 + rofs + r;
                float v = acc[mi][ni][r] + bv;
                if (RELU) v = v > 0.f ? v : 0.f;
                size_t o = (size_t)row * Nc + col;
                Cbf[o] = f2bf(v);
                if (STORE_F32) Cf[o] = v;
            }
        }
    }
}

// ---------------------------------------------------------------------------
// Distance + argmin: D[n,k] = |z_n|^2 + |c_k|^2 - 2 z.c ; u64 packed atomicMin
// Same tiling: 128 z-rows x 128 codebook entries per block.
// ---------------------------------------------------------------------------
__global__ __launch_bounds__(256)
void dist_argmin_kernel(const unsigned short* __restrict__ Zbf,
                        const unsigned short* __restrict__ CBbf,
                        const float* __restrict__ znorm,
                        const float* __restrict__ cnorm,
                        unsigned long long* __restrict__ best) {
    __shared__ __align__(16) unsigned short sA[128 * 40];
    __shared__ __align__(16) unsigned short sB[128 * 40];

    const int tid   = threadIdx.x;
    const int lane  = tid & 31;
    const int wave  = tid >> 5;
    const int waveM = wave >> 2;
    const int waveN = wave & 3;
    const int m0 = blockIdx.y * 128;   // z rows
    const int c0 = blockIdx.x * 128;   // codebook entries

    v8f acc[4][2] = {};

    const int arow = lane & 15;
    const int kbA  = (lane >= 16) ? 8 : 0;
    const int bcol = lane & 15;
    const int kbB  = (lane >= 16) ? 8 : 0;

    const int be = tid >> 1;
    const int bk = (tid & 1) * 16;

    for (int d0 = 0; d0 < D_Z; d0 += 32) {
        int idx = tid * 8;
        #pragma unroll
        for (int it = 0; it < 2; ++it) {
            int row = idx >> 5, kc = idx & 31;
            int4 v = *reinterpret_cast<const int4*>(
                &Zbf[(size_t)(m0 + row) * D_Z + d0 + kc]);
            *reinterpret_cast<int4*>(&sA[row * 40 + kc]) = v;
            idx += 2048;
        }
        {
            const unsigned short* g = &CBbf[(size_t)(c0 + be) * D_Z + d0 + bk];
            int4 v0 = *reinterpret_cast<const int4*>(g);
            int4 v1 = *reinterpret_cast<const int4*>(g + 8);
            *reinterpret_cast<int4*>(&sB[be * 40 + bk])     = v0;
            *reinterpret_cast<int4*>(&sB[be * 40 + bk + 8]) = v1;
        }
        __syncthreads();

        v16bf af[4], bfr[2];
        #pragma unroll
        for (int mi = 0; mi < 4; ++mi)
            af[mi] = load_fragA(&sA[(waveM * 64 + mi * 16 + arow) * 40 + kbA]);
        #pragma unroll
        for (int ni = 0; ni < 2; ++ni)
            bfr[ni] = load_fragB(&sB[(waveN * 32 + ni * 16 + bcol) * 40 + kbB]);

        #pragma unroll
        for (int mi = 0; mi < 4; ++mi)
            #pragma unroll
            for (int ni = 0; ni < 2; ++ni)
                acc[mi][ni] = __builtin_amdgcn_wmma_f32_16x16x32_bf16(
                    false, af[mi], false, bfr[ni], (short)0, acc[mi][ni],
                    false, false);
        __syncthreads();
    }

    const int rofs = (lane >= 16) ? 8 : 0;
    #pragma unroll
    for (int mi = 0; mi < 4; ++mi) {
        unsigned long long pk[8];
        #pragma unroll
        for (int r = 0; r < 8; ++r) pk[r] = ~0ULL;
        #pragma unroll
        for (int ni = 0; ni < 2; ++ni) {
            int col = c0 + waveN * 32 + ni * 16 + (lane & 15);
            float cn = cnorm[col];
            #pragma unroll
            for (int r = 0; r < 8; ++r) {
                int row = m0 + waveM * 64 + mi * 16 + rofs + r;
                float d = znorm[row] + cn - 2.0f * acc[mi][ni][r];
                if (d < 0.f) d = 0.f;   // keep f32 bit pattern order-monotone
                union { float f; unsigned u; } b; b.f = d;
                unsigned long long p =
                    ((unsigned long long)b.u << 32) | (unsigned)col;
                pk[r] = p < pk[r] ? p : pk[r];
            }
        }
        // min across the 16 lanes of each half (cols), then one atomic per row
        #pragma unroll
        for (int r = 0; r < 8; ++r) {
            unsigned long long p = pk[r];
            #pragma unroll
            for (int m = 1; m <= 8; m <<= 1) {
                unsigned long long q = shflxor64(p, m);
                p = q < p ? q : p;
            }
            if ((lane & 15) == 0) {
                int row = m0 + waveM * 64 + mi * 16 + rofs + r;
                atomicMin(&best[row], p);
            }
        }
    }
}

// ---------------------------------------------------------------------------
// helpers: row norms, init, gather (+ exact f32 min-dist), loss reduce
// ---------------------------------------------------------------------------
__global__ void rownorm_kernel(const float* __restrict__ X,
                               float* __restrict__ out, int rows, int cols) {
    int r = blockIdx.x * blockDim.x + threadIdx.x;
    if (r >= rows) return;
    const float4* p = reinterpret_cast<const float4*>(X + (size_t)r * cols);
    float s = 0.f;
    for (int i = 0; i < cols / 4; ++i) {
        float4 v = p[i];
        s += v.x * v.x + v.y * v.y + v.z * v.z + v.w * v.w;
    }
    out[r] = s;
}

__global__ void init_best_kernel(unsigned long long* best, int n) {
    int i = blockIdx.x * blockDim.x + threadIdx.x;
    if (i < n) best[i] = ~0ULL;
}

// 64 threads/block; copies codebook[k*] to out and recomputes the min
// distance exactly in f32: sum((z - c)^2)
__global__ __launch_bounds__(64)
void gather_kernel(const unsigned long long* __restrict__ best,
                   const float* __restrict__ codebook,
                   const float* __restrict__ Zf,
                   float* __restrict__ outQ,
                   float* __restrict__ minval) {
    __shared__ float red[64];
    int row = blockIdx.x;
    unsigned long long p = best[row];
    int k = (int)(p & 0xFFFFFFFFu);
    const float4* src = reinterpret_cast<const float4*>(codebook + (size_t)k * D_Z);
    const float4* zr  = reinterpret_cast<const float4*>(Zf + (size_t)row * D_Z);
    float4*       dst = reinterpret_cast<float4*>(outQ + (size_t)row * D_Z);
    int t = threadIdx.x;                  // 64 threads x float4 = 256 floats
    float4 c = src[t];
    float4 z = zr[t];
    dst[t] = c;
    float dx = z.x - c.x, dy = z.y - c.y, dz = z.z - c.z, dw = z.w - c.w;
    red[t] = dx * dx + dy * dy + dz * dz + dw * dw;
    __syncthreads();
    for (int w = 32; w > 0; w >>= 1) {
        if (t < w) red[t] += red[t + w];
        __syncthreads();
    }
    if (t == 0) minval[row] = red[0];
}

__global__ void loss_kernel(const float* __restrict__ minval,
                            float* __restrict__ out, int n) {
    __shared__ float red[256];
    float s = 0.f;
    for (int i = threadIdx.x; i < n; i += 256) s += minval[i];
    red[threadIdx.x] = s;
    __syncthreads();
    for (int w = 128; w > 0; w >>= 1) {
        if ((int)threadIdx.x < w) red[threadIdx.x] += red[threadIdx.x + w];
        __syncthreads();
    }
    if (threadIdx.x == 0) out[0] = 2.0f * red[0];  // z_loss + c_loss
}

// ---------------------------------------------------------------------------
extern "C" void kernel_launch(void* const* d_in, const int* in_sizes, int n_in,
                              void* d_out, int out_size, void* d_ws, size_t ws_size,
                              hipStream_t stream) {
    const float* x  = (const float*)d_in[0];
    const float* W1 = (const float*)d_in[1];
    const float* b1 = (const float*)d_in[2];
    const float* W2 = (const float*)d_in[3];
    const float* b2 = (const float*)d_in[4];
    const float* W3 = (const float*)d_in[5];
    const float* b3 = (const float*)d_in[6];
    const float* cb = (const float*)d_in[7];

    float* outQ    = (float*)d_out;                       // [N, D_Z]
    float* outLoss = outQ + (size_t)N_ROWS * D_Z;         // scalar

    char* ws = (char*)d_ws;
    auto take = [&](size_t bytes) {
        char* p = ws;
        ws += (bytes + 255) & ~(size_t)255;
        return p;
    };
    unsigned short* Xbf   = (unsigned short*)take((size_t)N_ROWS * D_IN * 2);
    unsigned short* W1t   = (unsigned short*)take((size_t)D_IN * D_H * 2);   // [D_H, D_IN]
    unsigned short* W2t   = (unsigned short*)take((size_t)D_H * D_H * 2);    // [D_H, D_H]
    unsigned short* W3t   = (unsigned short*)take((size_t)D_H * D_Z * 2);    // [D_Z, D_H]
    unsigned short* CBbf  = (unsigned short*)take((size_t)K_CODES * D_Z * 2);
    unsigned short* H1bf  = (unsigned short*)take((size_t)N_ROWS * D_H * 2);
    unsigned short* H2bf  = (unsigned short*)take((size_t)N_ROWS * D_H * 2);
    float*          Zf    = (float*)take((size_t)N_ROWS * D_Z * 4);
    unsigned short* Zbf   = (unsigned short*)take((size_t)N_ROWS * D_Z * 2);
    float*          znorm = (float*)take((size_t)N_ROWS * 4);
    float*          cnorm = (float*)take((size_t)K_CODES * 4);
    unsigned long long* best = (unsigned long long*)take((size_t)N_ROWS * 8);
    float*          minval = (float*)take((size_t)N_ROWS * 4);

    auto cvt = [&](const float* src, unsigned short* dst, size_t n) {
        int n4 = (int)(n / 4);
        convert_f32_bf16<<<(n4 + 255) / 256, 256, 0, stream>>>(src, dst, n4);
    };
    cvt(x,  Xbf,  (size_t)N_ROWS * D_IN);
    cvt(cb, CBbf, (size_t)K_CODES * D_Z);

    // transpose-convert weights: W [Kd,Nc] f32 -> WT [Nc,Kd] bf16
    transpose_convert_kernel<<<dim3(D_H / 32, D_IN / 32), 256, 0, stream>>>(W1, W1t, D_IN, D_H);
    transpose_convert_kernel<<<dim3(D_H / 32, D_H / 32), 256, 0, stream>>>(W2, W2t, D_H, D_H);
    transpose_convert_kernel<<<dim3(D_Z / 32, D_H / 32), 256, 0, stream>>>(W3, W3t, D_H, D_Z);

    // layer 1: H1 = relu(x @ W1 + b1)
    gemm_bf16_kernel<true, false><<<dim3(D_H / 128, N_ROWS / 128), 256, 0, stream>>>(
        Xbf, W1t, b1, H1bf, nullptr, N_ROWS, D_IN, D_H);
    // layer 2: H2 = H1 @ W2 + b2
    gemm_bf16_kernel<false, false><<<dim3(D_H / 128, N_ROWS / 128), 256, 0, stream>>>(
        H1bf, W2t, b2, H2bf, nullptr, N_ROWS, D_H, D_H);
    // layer 3: Z = H2 @ W3 + b3 (bf16 + f32)
    gemm_bf16_kernel<false, true><<<dim3(D_Z / 128, N_ROWS / 128), 256, 0, stream>>>(
        H2bf, W3t, b3, Zbf, Zf, N_ROWS, D_H, D_Z);

    rownorm_kernel<<<(N_ROWS + 255) / 256, 256, 0, stream>>>(Zf, znorm, N_ROWS, D_Z);
    rownorm_kernel<<<(K_CODES + 255) / 256, 256, 0, stream>>>(cb, cnorm, K_CODES, D_Z);
    init_best_kernel<<<(N_ROWS + 255) / 256, 256, 0, stream>>>(best, N_ROWS);

    dist_argmin_kernel<<<dim3(K_CODES / 128, N_ROWS / 128), 256, 0, stream>>>(
        Zbf, CBbf, znorm, cnorm, best);

    gather_kernel<<<N_ROWS, 64, 0, stream>>>(best, cb, Zf, outQ, minval);
    loss_kernel<<<1, 256, 0, stream>>>(minval, outLoss, N_ROWS);
}
